// FusedMoE_50483045597480
// MI455X (gfx1250) — matrix operand order repaired and checked by reading, not compiled, not run on hardware
//
#include <hip/hip_runtime.h>
#include <math.h>

// Problem constants (fixed by the reference setup).
#define T_TOK 2048
#define HDIM  2048
#define NEXP  8
#define IDIM  5632
#define TWOI  11264

typedef __attribute__((ext_vector_type(16))) __bf16 v16bf;
typedef __attribute__((ext_vector_type(8)))  float  v8f;
typedef __attribute__((ext_vector_type(4)))  float  f32x4;
typedef __attribute__((ext_vector_type(8)))  __bf16 bf16x8;
typedef __attribute__((ext_vector_type(2)))  __bf16 bf16x2;
typedef __attribute__((ext_vector_type(8)))  unsigned int u32x8;

// ---------------------------------------------------------------------------
// Pack two fp32 into one dword of two bf16 with a single VALU op.
// Preferred: v_cvt_pk_bf16_f32 (proper convert). Fallback: v_perm_b32
// truncation (grab high halves of both floats; selector 0x07060302).
// ---------------------------------------------------------------------------
__device__ __forceinline__ unsigned int pk2(float x, float y) {
#if __has_builtin(__builtin_amdgcn_cvt_pk_bf16_f32)
  bf16x2 t = __builtin_amdgcn_cvt_pk_bf16_f32(x, y);
  unsigned int u;
  __builtin_memcpy(&u, &t, 4);
  return u;
#else
  unsigned int ux, uy;
  __builtin_memcpy(&ux, &x, 4);
  __builtin_memcpy(&uy, &y, 4);
  return __builtin_amdgcn_perm(uy, ux, 0x07060302u);
#endif
}

// Scalar fp32 -> bf16 (round-to-nearest-even) for the epilogue store.
__device__ __forceinline__ __bf16 f2bf(float f) {
  unsigned int u;
  __builtin_memcpy(&u, &f, 4);
  u += 0x7fffu + ((u >> 16) & 1u);
  unsigned short s = (unsigned short)(u >> 16);
  __bf16 b;
  __builtin_memcpy(&b, &s, 2);
  return b;
}

// 16 fp32 -> v16bf fragment via 8 packed dwords (lands directly in the
// 8 fragment VGPRs, no element-wise v_mov_b16).
__device__ __forceinline__ v16bf cvt16(f32x4 a, f32x4 b, f32x4 c, f32x4 d) {
  u32x8 p;
  p[0] = pk2(a[0], a[1]);
  p[1] = pk2(a[2], a[3]);
  p[2] = pk2(b[0], b[1]);
  p[3] = pk2(b[2], b[3]);
  p[4] = pk2(c[0], c[1]);
  p[5] = pk2(c[2], c[3]);
  p[6] = pk2(d[0], d[1]);
  p[7] = pk2(d[2], d[3]);
  v16bf r;
  __builtin_memcpy(&r, &p, 32);
  return r;
}

// ---------------------------------------------------------------------------
// Kernel 1: zero output accumulator + expert counters.
// ---------------------------------------------------------------------------
__global__ void moe_zero_kernel(float* __restrict__ out, int n, int* __restrict__ counts) {
  int i = blockIdx.x * blockDim.x + threadIdx.x;
  if (i < n) out[i] = 0.0f;
  if (i < NEXP) counts[i] = 0;
}

// ---------------------------------------------------------------------------
// Kernel 2: softmax top-2 router + scatter into per-expert token lists.
// softmax top-2 renormalized == pairwise softmax of the two top logits.
// ---------------------------------------------------------------------------
__global__ void moe_router_kernel(const float* __restrict__ logits,
                                  int* __restrict__ counts,
                                  int* __restrict__ list,
                                  float* __restrict__ gates) {
  int t = blockIdx.x * blockDim.x + threadIdx.x;
  if (t >= T_TOK) return;
  const float* lr = logits + (size_t)t * NEXP;
  float v0 = -3.4e38f, v1 = -3.4e38f;
  int i0 = 0, i1 = 0;
#pragma unroll
  for (int e = 0; e < NEXP; ++e) {
    float v = lr[e];
    if (v > v0) { v1 = v0; i1 = i0; v0 = v; i0 = e; }
    else if (v > v1) { v1 = v; i1 = e; }
  }
  float eb = __expf(v1 - v0);
  float g0 = 1.0f / (1.0f + eb);
  float g1 = 1.0f - g0;
  int p0 = atomicAdd(&counts[i0], 1);
  list[i0 * T_TOK + p0]  = t;
  gates[i0 * T_TOK + p0] = g0;
  int p1 = atomicAdd(&counts[i1], 1);
  list[i1 * T_TOK + p1]  = t;
  gates[i1 * T_TOK + p1] = g1;
}

// ---------------------------------------------------------------------------
// Kernel 3: gemm1 for one expert.  Block = 8 waves = 2(M)x4(N) 16x16 tiles ->
// 32(M) x 64(N) tile.  Computes BOTH the gate half (cols n) and up half
// (cols n + I) of gu, applies silu(g)*u in registers, writes bf16 h.
// ---------------------------------------------------------------------------
__global__ __launch_bounds__(256) void moe_gemm1_kernel(
    const float* __restrict__ hs, const float* __restrict__ w13,
    const int* __restrict__ counts, const int* __restrict__ list,
    unsigned short* __restrict__ hbuf_raw, int e) {
  const int count = counts[e];
  const int mtile = blockIdx.x;
  const int ntile = blockIdx.y;
  if (mtile * 32 >= count) return;           // block-uniform exit

  __bf16* hbuf = (__bf16*)hbuf_raw;
  const int wid  = threadIdx.x >> 5;
  const int lane = threadIdx.x & 31;
  const int waveM = wid & 1;
  const int waveN = wid >> 1;
  const int l15 = lane & 15;
  const int kh  = lane >> 4;                 // which 16-lane half

  // A: gathered token row (contiguous k order used here, same as B: lanes0-15 K 0..15,
  // lanes16-31 K 16..31).
  const int slot    = mtile * 32 + waveM * 16 + l15;
  const int slot_cl = (slot < count) ? slot : (count - 1);
  const int token   = list[e * T_TOK + slot_cl];
  const float* a0p = hs + (size_t)token * HDIM + kh * 16;
  const float* a1p = a0p + 8;

  // B: w13 row = output column; K contiguous. (k order used here, same as A: lanes0-15
  // K 0..15, lanes16-31 K 16..31, N = lane%16.)
  const int ncol = ntile * 64 + waveN * 16 + l15;            // in [0, IDIM)
  const float* bg = w13 + ((size_t)e * TWOI + ncol) * HDIM + kh * 16;
  const float* bu = bg + (size_t)IDIM * HDIM;                // up-proj half

  v8f accg = {};
  v8f accu = {};
  for (int kb = 0; kb < HDIM; kb += 32) {
    __builtin_prefetch(bg + kb + 256, 0, 1);
    __builtin_prefetch(bu + kb + 256, 0, 1);
    f32x4 a0 = *(const f32x4*)(a0p + kb);
    f32x4 a1 = *(const f32x4*)(a0p + kb + 4);
    f32x4 a2 = *(const f32x4*)(a1p + kb);
    f32x4 a3 = *(const f32x4*)(a1p + kb + 4);
    v16bf A = cvt16(a0, a1, a2, a3);

    f32x4 g0 = *(const f32x4*)(bg + kb);
    f32x4 g1 = *(const f32x4*)(bg + kb + 4);
    f32x4 g2 = *(const f32x4*)(bg + kb + 8);
    f32x4 g3 = *(const f32x4*)(bg + kb + 12);
    v16bf Bg = cvt16(g0, g1, g2, g3);

    f32x4 u0 = *(const f32x4*)(bu + kb);
    f32x4 u1 = *(const f32x4*)(bu + kb + 4);
    f32x4 u2 = *(const f32x4*)(bu + kb + 8);
    f32x4 u3 = *(const f32x4*)(bu + kb + 12);
    v16bf Bu = cvt16(u0, u1, u2, u3);

    accg = __builtin_amdgcn_wmma_f32_16x16x32_bf16(
        false, A, false, Bg, (short)0, accg, false, false);
    accu = __builtin_amdgcn_wmma_f32_16x16x32_bf16(
        false, A, false, Bu, (short)0, accu, false, false);
  }

  // D layout: VGPR r, lanes0-15 -> M=r, lanes16-31 -> M=r+8; N = lane%16.
#pragma unroll
  for (int r = 0; r < 8; ++r) {
    int srow = mtile * 32 + waveM * 16 + r + kh * 8;
    if (srow < count) {
      float g = accg[r];
      float u = accu[r];
      // silu(g)*u with fast reciprocal (v_rcp_f32) instead of IEEE divide.
      float hv = g * __builtin_amdgcn_rcpf(1.0f + __expf(-g)) * u;
      hbuf[(size_t)srow * IDIM + ncol] = f2bf(hv);
    }
  }
}

// ---------------------------------------------------------------------------
// Kernel 4: gemm2 for one expert: out[token] += gate * (h @ w2[e]^T).
// Each (srow, ncol) is touched by exactly one lane, and expert kernels are
// stream-serialized, so the += needs no atomics.
// ---------------------------------------------------------------------------
__global__ __launch_bounds__(256) void moe_gemm2_kernel(
    const unsigned short* __restrict__ hbuf_raw, const float* __restrict__ w2,
    const int* __restrict__ counts, const int* __restrict__ list,
    const float* __restrict__ gates, float* __restrict__ out, int e) {
  const int count = counts[e];
  const int mtile = blockIdx.x;
  const int ntile = blockIdx.y;
  if (mtile * 32 >= count) return;

  const __bf16* hbuf = (const __bf16*)hbuf_raw;
  const int wid  = threadIdx.x >> 5;
  const int lane = threadIdx.x & 31;
  const int waveM = wid & 1;
  const int waveN = wid >> 1;
  const int l15 = lane & 15;
  const int kh  = lane >> 4;

  const int slot    = mtile * 32 + waveM * 16 + l15;
  const int slot_cl = (slot < count) ? slot : (count - 1);
  const __bf16* ap = hbuf + (size_t)slot_cl * IDIM + kh * 16;

  const int ncol = ntile * 64 + waveN * 16 + l15;            // in [0, HDIM)
  const float* bp = w2 + ((size_t)e * HDIM + ncol) * IDIM + kh * 16;

  v8f acc = {};
  for (int kb = 0; kb < IDIM; kb += 32) {
    __builtin_prefetch(bp + kb + 256, 0, 1);
    bf16x8 a0 = *(const bf16x8*)(ap + kb);
    bf16x8 a1 = *(const bf16x8*)(ap + kb + 8);
    // Concatenate the two halves; register-coalesced, no v_mov_b16 chain.
    v16bf A = __builtin_shufflevector(a0, a1, 0, 1, 2, 3, 4, 5, 6, 7,
                                      8, 9, 10, 11, 12, 13, 14, 15);

    f32x4 b0 = *(const f32x4*)(bp + kb);
    f32x4 b1 = *(const f32x4*)(bp + kb + 4);
    f32x4 b2 = *(const f32x4*)(bp + kb + 8);
    f32x4 b3 = *(const f32x4*)(bp + kb + 12);
    v16bf B = cvt16(b0, b1, b2, b3);

    acc = __builtin_amdgcn_wmma_f32_16x16x32_bf16(
        false, A, false, B, (short)0, acc, false, false);
  }

#pragma unroll
  for (int r = 0; r < 8; ++r) {
    int srow = mtile * 32 + waveM * 16 + r + kh * 8;
    if (srow < count) {
      int   token = list[e * T_TOK + srow];
      float gate  = gates[e * T_TOK + srow];
      float* o = out + (size_t)token * HDIM + ncol;
      *o += gate * acc[r];
    }
  }
}

// ---------------------------------------------------------------------------
// Host launcher.
// Workspace layout (d_ws): [counts:8 int][list:8*2048 int][gates:8*2048 float]
//                          [hbuf: 2048*5632 bf16]  (~23.2 MB total)
// ---------------------------------------------------------------------------
extern "C" void kernel_launch(void* const* d_in, const int* in_sizes, int n_in,
                              void* d_out, int out_size, void* d_ws, size_t ws_size,
                              hipStream_t stream) {
  const float* hs     = (const float*)d_in[0];   // (T, H) fp32
  const float* logits = (const float*)d_in[1];   // (T, E) fp32
  const float* w13    = (const float*)d_in[2];   // (E, 2I, H) fp32
  const float* w2     = (const float*)d_in[3];   // (E, H, I) fp32
  float* out = (float*)d_out;                    // (T, H) fp32

  char* ws = (char*)d_ws;
  int*   counts = (int*)ws;
  int*   list   = counts + NEXP;
  float* gates  = (float*)(list + NEXP * T_TOK);
  unsigned short* hbuf = (unsigned short*)(gates + NEXP * T_TOK);

  // 1) zero output accumulator + counters
  {
    int n = T_TOK * HDIM;
    moe_zero_kernel<<<(n + 255) / 256, 256, 0, stream>>>(out, n, counts);
  }
  // 2) router + scatter
  moe_router_kernel<<<(T_TOK + 255) / 256, 256, 0, stream>>>(logits, counts, list, gates);

  // 3/4) per-expert fused gemm1(silu) -> gemm2(scatter-add), sequential on
  // the stream so the shared h buffer is safely reused.
  for (int e = 0; e < NEXP; ++e) {
    dim3 g1(T_TOK / 32, IDIM / 64);   // 64 x 88 blocks, early-exit on count
    moe_gemm1_kernel<<<g1, 256, 0, stream>>>(hs, w13, counts, list, hbuf, e);
    dim3 g2(T_TOK / 32, HDIM / 64);   // 64 x 32 blocks
    moe_gemm2_kernel<<<g2, 256, 0, stream>>>(hbuf, w2, counts, list, gates, out, e);
  }
}
